// StreamTimeAttn_1632087573190
// MI455X (gfx1250) — matrix operand order repaired and checked by reading, not compiled, not run on hardware
//
#include <hip/hip_runtime.h>
#include <cstdint>
#include <cstddef>

typedef __bf16 bf16_t;
typedef __attribute__((ext_vector_type(16))) __bf16 v16bf;
typedef __attribute__((ext_vector_type(8)))  __bf16 v8bf;
typedef __attribute__((ext_vector_type(8)))  float  v8f;

union FragBF { v16bf v; v8bf h[2]; };

#define BATCH 16
#define CH 2
#define TDIM 16
#define FDIM 256
#define TPP 4
#define FPP 4
#define MF 64
#define TIC 4
#define SEQ 256
#define LEFT 768
#define KTOT 1024
#define MT 16
#define DMODEL 512
#define NH 8
#define NG 4
#define HD 64
#define NL 4
#define FFH 2048
#define PATCH 32

// ---------------------------------------------------------------------------
// Register-blocked NT GEMM: C[M,N] = alpha * (A[M,K] @ Bt[N,K]^T + bias) + resid
// A row-major (lda), Bt row-major N x K (ldb).
// 8 waves/block arranged 2(M) x 4(N); each wave computes a 2x4 grid of 16x16
// tiles (32 rows x 64 cols), reusing A fragments across N tiles and B
// fragments across M tiles: per k-step 6 fragment loads feed 8 WMMAs.
// Block tile = 64 x 256. Requires M % 64 == 0 (true for all uses: M = 4096).
// Fragments follow CDNA5 16-bit WMMA layouts: A and B both interleaved K 0-7/16-23
// (lanes<16) & 8-15/24-31.
// ---------------------------------------------------------------------------
__global__ __launch_bounds__(256)
void gemm_nt_kernel(const bf16_t* __restrict__ A, int lda,
                    const bf16_t* __restrict__ Bt, int ldb,
                    const float* __restrict__ bias,
                    const float* __restrict__ resid,
                    const float* __restrict__ alpha_ptr, int alpha_idx,
                    float* __restrict__ Cf, bf16_t* __restrict__ Cbf,
                    int ldc, int M, int N, int K)
{
  int tid  = threadIdx.x;
  int wave = tid >> 5, lane = tid & 31;
  int wm = wave >> 2;                 // 0..1
  int wn = wave & 3;                  // 0..3
  int m_base = blockIdx.y * 64 + wm * 32;       // wave covers rows [m_base, m_base+32)
  int n_base = blockIdx.x * 256 + wn * 64;      // wave covers cols [n_base, n_base+64)
  int mrow = lane & 15;
  int ncol = lane & 15;
  int koff = (lane < 16) ? 0 : 8;
  int kb   = (lane < 16) ? 0 : 16;

  const bf16_t* arow[2];
#pragma unroll
  for (int tm = 0; tm < 2; ++tm)
    arow[tm] = A + (size_t)(m_base + tm * 16 + mrow) * lda;

  const bf16_t* brow[4];
#pragma unroll
  for (int tn = 0; tn < 4; ++tn) {
    int nIdx = n_base + tn * 16 + ncol;
    if (nIdx > N - 1) nIdx = N - 1;            // clamp (garbage compute, store guarded)
    brow[tn] = Bt + (size_t)nIdx * ldb;
  }

  v8f acc[2][4] = {};
  for (int k0 = 0; k0 < K; k0 += 32) {
    FragBF a[2], b[4];
#pragma unroll
    for (int tm = 0; tm < 2; ++tm) {
      a[tm].h[0] = *reinterpret_cast<const v8bf*>(arow[tm] + k0 + koff);
      a[tm].h[1] = *reinterpret_cast<const v8bf*>(arow[tm] + k0 + 16 + koff);
    }
#pragma unroll
    for (int tn = 0; tn < 4; ++tn) {
      b[tn].h[0] = *reinterpret_cast<const v8bf*>(brow[tn] + k0 + koff);
      b[tn].h[1] = *reinterpret_cast<const v8bf*>(brow[tn] + k0 + 16 + koff);
    }
#pragma unroll
    for (int tm = 0; tm < 2; ++tm)
#pragma unroll
      for (int tn = 0; tn < 4; ++tn)
        acc[tm][tn] = __builtin_amdgcn_wmma_f32_16x16x32_bf16(
            false, a[tm].v, false, b[tn].v, (short)0, acc[tm][tn], false, false);
  }

  float alpha = alpha_ptr ? alpha_ptr[alpha_idx] : 1.0f;
#pragma unroll
  for (int tn = 0; tn < 4; ++tn) {
    int n = n_base + tn * 16 + ncol;
    if (n >= N) continue;
    float bi = bias ? bias[n] : 0.0f;
#pragma unroll
    for (int tm = 0; tm < 2; ++tm) {
#pragma unroll
      for (int v = 0; v < 8; ++v) {
        int m = m_base + tm * 16 + ((lane < 16) ? v : v + 8);
        size_t off = (size_t)m * ldc + n;
        float val = (acc[tm][tn][v] + bi) * alpha;
        if (resid) val += resid[off];
        if (Cf)  Cf[off]  = val;
        if (Cbf) Cbf[off] = (bf16_t)val;
      }
    }
  }
}

// ---------------------------------------------------------------------------
// Fused attention: per block (1 wave): 16 query rows of one (b, head).
// scores = Q K^T * 0.125 + relpos bias -> softmax -> P @ V.
// ---------------------------------------------------------------------------
__global__ __launch_bounds__(32)
void attn_kernel(const bf16_t* __restrict__ qbf,
                 const bf16_t* __restrict__ kfull,   // (B,G,KTOT,HD) bf16
                 const bf16_t* __restrict__ vfullt,  // (B,G,HD,KTOT) bf16
                 const float* __restrict__ rel_time_l,  // (2*MT-1, NH)
                 const float* __restrict__ rel_freq_l,  // (2*MF-1, NH)
                 bf16_t* __restrict__ obf)           // (B,SEQ,DMODEL)
{
  extern __shared__ char smem[];
  float*  sc = (float*)smem;                                   // 16*1024 f32
  bf16_t* pr = (bf16_t*)(smem + 16 * KTOT * 4);                // 16*1024 bf16
  float*  rt = (float*)(smem + 16 * KTOT * 4 + 16 * KTOT * 2); // 31
  float*  rf = rt + 32;                                        // 127

  int lane = threadIdx.x;
  int s0 = blockIdx.x * 16;
  int hh = blockIdx.y;
  int b  = blockIdx.z;
  int g  = hh >> 1;  // R = 2 query heads per kv group

  for (int i = lane; i < 2 * MT - 1; i += 32) rt[i] = rel_time_l[i * NH + hh];
  for (int i = lane; i < 2 * MF - 1; i += 32) rf[i] = rel_freq_l[i * NH + hh];
  __syncthreads();

  int mrow = lane & 15, ncol = lane & 15;
  int koff = (lane < 16) ? 0 : 8;
  int kb   = (lane < 16) ? 0 : 16;

  // Q fragments for the whole HD=64 (two k-steps of 32), held in registers
  const bf16_t* qrow = qbf + ((size_t)b * SEQ + s0 + mrow) * DMODEL + hh * HD;
  FragBF a0, a1;
  a0.h[0] = *(const v8bf*)(qrow + koff);
  a0.h[1] = *(const v8bf*)(qrow + 16 + koff);
  a1.h[0] = *(const v8bf*)(qrow + 32 + koff);
  a1.h[1] = *(const v8bf*)(qrow + 48 + koff);

  const bf16_t* kbase = kfull + ((size_t)b * NG + g) * (size_t)KTOT * HD;
  for (int n0 = 0; n0 < KTOT; n0 += 16) {
    const bf16_t* krow = kbase + (size_t)(n0 + ncol) * HD;
    FragBF b0, b1;
    b0.h[0] = *(const v8bf*)(krow + koff);
    b0.h[1] = *(const v8bf*)(krow + 16 + koff);
    b1.h[0] = *(const v8bf*)(krow + 32 + koff);
    b1.h[1] = *(const v8bf*)(krow + 48 + koff);
    v8f acc = {};
    acc = __builtin_amdgcn_wmma_f32_16x16x32_bf16(false, a0.v, false, b0.v,
                                                  (short)0, acc, false, false);
    acc = __builtin_amdgcn_wmma_f32_16x16x32_bf16(false, a1.v, false, b1.v,
                                                  (short)0, acc, false, false);
    int t  = n0 + ncol;
    int kt = t >> 6, kf = t & 63;
#pragma unroll
    for (int v = 0; v < 8; ++v) {
      int m = (lane < 16) ? v : v + 8;
      int s = s0 + m;
      int dt = (12 + (s >> 6)) - kt + (MT - 1);   // LEFT/MF = 12
      int df = (s & 63) - kf + (MF - 1);
      sc[m * KTOT + t] = acc[v] * 0.125f + rt[dt] + rf[df];
    }
  }
  __syncthreads();

  // softmax per row, write bf16 probs
  for (int m = 0; m < 16; ++m) {
    float mx = -1e30f;
    for (int i = lane; i < KTOT; i += 32) mx = fmaxf(mx, sc[m * KTOT + i]);
    for (int o = 16; o > 0; o >>= 1) mx = fmaxf(mx, __shfl_xor(mx, o, 32));
    float sum = 0.f;
    for (int i = lane; i < KTOT; i += 32) {
      float e = __expf(sc[m * KTOT + i] - mx);
      sc[m * KTOT + i] = e;
      sum += e;
    }
    for (int o = 16; o > 0; o >>= 1) sum += __shfl_xor(sum, o, 32);
    float inv = 1.0f / sum;
    for (int i = lane; i < KTOT; i += 32)
      pr[m * KTOT + i] = (bf16_t)(sc[m * KTOT + i] * inv);
  }
  __syncthreads();

  // O = P(16x1024) @ V^T ; vfullt rows are hd (NT layout)
  const bf16_t* vbase = vfullt + ((size_t)b * NG + g) * (size_t)HD * KTOT;
  v8f oacc[4] = {};
  for (int k0 = 0; k0 < KTOT; k0 += 32) {
    FragBF a;
    a.h[0] = *(const v8bf*)(pr + mrow * KTOT + k0 + koff);
    a.h[1] = *(const v8bf*)(pr + mrow * KTOT + k0 + 16 + koff);
#pragma unroll
    for (int nt = 0; nt < 4; ++nt) {
      const bf16_t* vrow = vbase + (size_t)(nt * 16 + ncol) * KTOT;
      FragBF bb;
      bb.h[0] = *(const v8bf*)(vrow + k0 + koff);
      bb.h[1] = *(const v8bf*)(vrow + k0 + 16 + koff);
      oacc[nt] = __builtin_amdgcn_wmma_f32_16x16x32_bf16(false, a.v, false, bb.v,
                                                         (short)0, oacc[nt], false, false);
    }
  }
#pragma unroll
  for (int nt = 0; nt < 4; ++nt)
#pragma unroll
    for (int v = 0; v < 8; ++v) {
      int m = (lane < 16) ? v : v + 8;
      int s = s0 + m;
      int d = nt * 16 + ncol;
      obf[((size_t)b * SEQ + s) * DMODEL + hh * HD + d] = (bf16_t)oacc[nt][v];
    }
}

// ---------------------------------------------------------------------------
// LayerNorm over D=512 (one row per 256-thread block), bf16 output
// ---------------------------------------------------------------------------
__global__ __launch_bounds__(256)
void ln_kernel(const float* __restrict__ x, const float* __restrict__ gamma,
               const float* __restrict__ beta, bf16_t* __restrict__ out)
{
  __shared__ float red[256];
  int row = blockIdx.x;
  int tid = threadIdx.x;
  const float* xr = x + (size_t)row * DMODEL;
  float a = xr[tid], b = xr[tid + 256];
  red[tid] = a + b;
  __syncthreads();
  for (int o = 128; o > 0; o >>= 1) { if (tid < o) red[tid] += red[tid + o]; __syncthreads(); }
  float mean = red[0] * (1.0f / DMODEL);
  __syncthreads();
  float da = a - mean, db = b - mean;
  red[tid] = da * da + db * db;
  __syncthreads();
  for (int o = 128; o > 0; o >>= 1) { if (tid < o) red[tid] += red[tid + o]; __syncthreads(); }
  float rstd = rsqrtf(red[0] * (1.0f / DMODEL) + 1e-5f);
  bf16_t* orow = out + (size_t)row * DMODEL;
  orow[tid]       = (bf16_t)(da * rstd * gamma[tid] + beta[tid]);
  orow[tid + 256] = (bf16_t)(db * rstd * gamma[tid + 256] + beta[tid + 256]);
}

// Transpose + f32->bf16: dst[n*K+k] = src[k*N+n]
__global__ void transpose_cvt_kernel(const float* __restrict__ src,
                                     bf16_t* __restrict__ dst, int K, int N)
{
  int idx = blockIdx.x * 256 + threadIdx.x;
  if (idx >= K * N) return;
  int n = idx / K, k = idx - n * K;
  dst[idx] = (bf16_t)src[(size_t)k * N + n];
}

// (B,C,T,F) -> patch tokens (B,S,32) bf16
__global__ void patchify_kernel(const float* __restrict__ x, bf16_t* __restrict__ P)
{
  int idx = blockIdx.x * 256 + threadIdx.x;  // B*S*32 = 131072
  int j = idx & 31;
  int s = (idx >> 5) & (SEQ - 1);
  int b = idx >> 13;
  int tic = s >> 6, mf = s & 63;
  int c = j & 1, fp = (j >> 1) & 3, tp = j >> 3;
  int t = tic * TPP + tp, f = mf * FPP + fp;
  P[idx] = (bf16_t)x[(((size_t)b * CH + c) * TDIM + t) * FDIM + f];
}

// y (B,S,32) f32 -> out (B,C,T,F)
__global__ void unpatchify_kernel(const float* __restrict__ y, float* __restrict__ out)
{
  int idx = blockIdx.x * 256 + threadIdx.x;  // B*C*T*F = 131072
  int f = idx & 255;
  int t = (idx >> 8) & 15;
  int c = (idx >> 12) & 1;
  int b = idx >> 13;
  int tic = t >> 2, tp = t & 3, mf = f >> 2, fp = f & 3;
  int s = tic * MF + mf;
  int j = (tp * FPP + fp) * CH + c;
  out[idx] = y[((size_t)b * SEQ + s) * PATCH + j];
}

// Build bf16 K (token-major) and V^T (hd-major) working buffers from cache +
// new projections; emit rotated f32 caches to d_out.
__global__ void kvpack_kernel(const float* __restrict__ cache_k_l,
                              const float* __restrict__ cache_v_l,
                              const float* __restrict__ knew,  // (B,S,G*HD)
                              const float* __restrict__ vnew,
                              bf16_t* __restrict__ kfull,      // (B,G,KTOT,HD)
                              bf16_t* __restrict__ vfullt,     // (B,G,HD,KTOT)
                              float* __restrict__ nck, float* __restrict__ ncv)
{
  int idx = blockIdx.x * 256 + threadIdx.x;   // B*G*KTOT*HD = 4194304
  int d   = idx & 63;
  int tok = (idx >> 6) & (KTOT - 1);
  int g   = (idx >> 16) & 3;
  int b   = idx >> 18;
  float kv_, vv_;
  if (tok < LEFT) {
    size_t co = (((size_t)b * NG + g) * LEFT + tok) * HD + d;
    kv_ = cache_k_l[co]; vv_ = cache_v_l[co];
  } else {
    int s = tok - LEFT;
    size_t no = ((size_t)b * SEQ + s) * (NG * HD) + g * HD + d;
    kv_ = knew[no]; vv_ = vnew[no];
  }
  kfull[idx] = (bf16_t)kv_;
  vfullt[(((size_t)b * NG + g) * HD + d) * KTOT + tok] = (bf16_t)vv_;
  if (tok >= SEQ) {
    size_t oo = (((size_t)b * NG + g) * LEFT + (tok - SEQ)) * HD + d;
    nck[oo] = kv_; ncv[oo] = vv_;
  }
}

// ff = silu(g1) * g3, bf16 out
__global__ void swiglu_kernel(const float* __restrict__ g1,
                              const float* __restrict__ g3,
                              bf16_t* __restrict__ ff, int n)
{
  int idx = blockIdx.x * 256 + threadIdx.x;
  if (idx >= n) return;
  float a = g1[idx];
  float s = a / (1.0f + __expf(-a));
  ff[idx] = (bf16_t)(s * g3[idx]);
}

// ---------------------------------------------------------------------------
extern "C" void kernel_launch(void* const* d_in, const int* in_sizes, int n_in,
                              void* d_out, int out_size, void* d_ws, size_t ws_size,
                              hipStream_t stream)
{
  const float* x        = (const float*)d_in[0];
  const float* cache_k  = (const float*)d_in[1];
  const float* cache_v  = (const float*)d_in[2];
  const float* in_W     = (const float*)d_in[3];
  const float* in_b     = (const float*)d_in[4];
  const float* ln1_g    = (const float*)d_in[5];
  const float* ln1_b    = (const float*)d_in[6];
  const float* Wq       = (const float*)d_in[7];
  const float* bq       = (const float*)d_in[8];
  const float* Wk       = (const float*)d_in[9];
  const float* bk       = (const float*)d_in[10];
  const float* Wv       = (const float*)d_in[11];
  const float* bv       = (const float*)d_in[12];
  const float* Wo       = (const float*)d_in[13];
  const float* bo       = (const float*)d_in[14];
  const float* rel_time = (const float*)d_in[15];
  const float* rel_freq = (const float*)d_in[16];
  const float* res_a    = (const float*)d_in[17];
  const float* ln2_g    = (const float*)d_in[18];
  const float* ln2_b    = (const float*)d_in[19];
  const float* W1       = (const float*)d_in[20];
  const float* b1       = (const float*)d_in[21];
  const float* W3       = (const float*)d_in[22];
  const float* b3       = (const float*)d_in[23];
  const float* W2       = (const float*)d_in[24];
  const float* b2       = (const float*)d_in[25];
  const float* res_f    = (const float*)d_in[26];
  const float* out_W    = (const float*)d_in[27];
  const float* out_b    = (const float*)d_in[28];

  float* y_out   = (float*)d_out;
  float* nck_out = y_out + (size_t)BATCH * CH * TDIM * FDIM;
  float* ncv_out = nck_out + (size_t)NL * BATCH * NG * LEFT * HD;

  // workspace carve-out
  char* w = (char*)d_ws;
  auto alloc = [&](size_t bytes) -> char* {
    char* p = w; w += (bytes + 255) & ~(size_t)255; return p;
  };
  const int Mrows = BATCH * SEQ;  // 4096
  float*  h      = (float*) alloc((size_t)Mrows * DMODEL * 4);
  bf16_t* xinb   = (bf16_t*)alloc((size_t)Mrows * DMODEL * 2);
  bf16_t* Pbf    = (bf16_t*)alloc((size_t)Mrows * PATCH * 2);
  bf16_t* qbf    = (bf16_t*)alloc((size_t)Mrows * DMODEL * 2);
  float*  kf32   = (float*) alloc((size_t)Mrows * NG * HD * 4);
  float*  vf32   = (float*) alloc((size_t)Mrows * NG * HD * 4);
  bf16_t* kfull  = (bf16_t*)alloc((size_t)BATCH * NG * KTOT * HD * 2);
  bf16_t* vfullt = (bf16_t*)alloc((size_t)BATCH * NG * KTOT * HD * 2);
  bf16_t* obf    = (bf16_t*)alloc((size_t)Mrows * DMODEL * 2);
  float*  gbuf1  = (float*) alloc((size_t)Mrows * FFH * 4);
  float*  gbuf3  = (float*) alloc((size_t)Mrows * FFH * 4);
  bf16_t* ffbf   = (bf16_t*)alloc((size_t)Mrows * FFH * 2);
  bf16_t* hbf    = (bf16_t*)alloc((size_t)Mrows * DMODEL * 2);
  float*  ytmp   = (float*) alloc((size_t)Mrows * PATCH * 4);
  bf16_t* inWT   = (bf16_t*)alloc((size_t)PATCH * DMODEL * 2);
  bf16_t* outWT  = (bf16_t*)alloc((size_t)DMODEL * PATCH * 2);
  bf16_t *WqT[NL], *WkT[NL], *WvT[NL], *WoT[NL], *W1T[NL], *W3T[NL], *W2T[NL];
  for (int l = 0; l < NL; ++l) {
    WqT[l] = (bf16_t*)alloc((size_t)DMODEL * DMODEL * 2);
    WkT[l] = (bf16_t*)alloc((size_t)DMODEL * NG * HD * 2);
    WvT[l] = (bf16_t*)alloc((size_t)DMODEL * NG * HD * 2);
    WoT[l] = (bf16_t*)alloc((size_t)DMODEL * DMODEL * 2);
    W1T[l] = (bf16_t*)alloc((size_t)DMODEL * FFH * 2);
    W3T[l] = (bf16_t*)alloc((size_t)DMODEL * FFH * 2);
    W2T[l] = (bf16_t*)alloc((size_t)FFH * DMODEL * 2);
  }

  auto tcvt = [&](const float* src, bf16_t* dst, int K, int N) {
    int tot = K * N;
    transpose_cvt_kernel<<<(tot + 255) / 256, 256, 0, stream>>>(src, dst, K, N);
  };
  tcvt(in_W, inWT, PATCH, DMODEL);
  tcvt(out_W, outWT, DMODEL, PATCH);
  for (int l = 0; l < NL; ++l) {
    tcvt(Wq + (size_t)l * DMODEL * DMODEL, WqT[l], DMODEL, DMODEL);
    tcvt(Wk + (size_t)l * DMODEL * NG * HD, WkT[l], DMODEL, NG * HD);
    tcvt(Wv + (size_t)l * DMODEL * NG * HD, WvT[l], DMODEL, NG * HD);
    tcvt(Wo + (size_t)l * DMODEL * DMODEL, WoT[l], DMODEL, DMODEL);
    tcvt(W1 + (size_t)l * DMODEL * FFH, W1T[l], DMODEL, FFH);
    tcvt(W3 + (size_t)l * DMODEL * FFH, W3T[l], DMODEL, FFH);
    tcvt(W2 + (size_t)l * FFH * DMODEL, W2T[l], FFH, DMODEL);
  }

  patchify_kernel<<<(Mrows * PATCH) / 256, 256, 0, stream>>>(x, Pbf);

  auto gemm = [&](const bf16_t* A, int lda, const bf16_t* Bt, int ldb,
                  const float* bias, const float* resid,
                  const float* aptr, int aidx,
                  float* Cf, bf16_t* Cbf, int ldc, int M, int N, int K) {
    dim3 grid((N + 255) / 256, M / 64);
    gemm_nt_kernel<<<grid, 256, 0, stream>>>(A, lda, Bt, ldb, bias, resid,
                                             aptr, aidx, Cf, Cbf, ldc, M, N, K);
  };

  // h = patchify(x) @ in_W + in_b
  gemm(Pbf, PATCH, inWT, PATCH, in_b, nullptr, nullptr, 0,
       h, nullptr, DMODEL, Mrows, DMODEL, PATCH);

  size_t attn_lds = (size_t)16 * KTOT * 4 + (size_t)16 * KTOT * 2 + 256 * 4;

  for (int l = 0; l < NL; ++l) {
    ln_kernel<<<Mrows, 256, 0, stream>>>(h, ln1_g + l * DMODEL, ln1_b + l * DMODEL, xinb);
    gemm(xinb, DMODEL, WqT[l], DMODEL, bq + l * DMODEL, nullptr, nullptr, 0,
         nullptr, qbf, DMODEL, Mrows, DMODEL, DMODEL);
    gemm(xinb, DMODEL, WkT[l], DMODEL, bk + l * NG * HD, nullptr, nullptr, 0,
         kf32, nullptr, NG * HD, Mrows, NG * HD, DMODEL);
    gemm(xinb, DMODEL, WvT[l], DMODEL, bv + l * NG * HD, nullptr, nullptr, 0,
         vf32, nullptr, NG * HD, Mrows, NG * HD, DMODEL);

    kvpack_kernel<<<(BATCH * NG * KTOT * HD) / 256, 256, 0, stream>>>(
        cache_k + (size_t)l * BATCH * NG * LEFT * HD,
        cache_v + (size_t)l * BATCH * NG * LEFT * HD,
        kf32, vf32, kfull, vfullt,
        nck_out + (size_t)l * BATCH * NG * LEFT * HD,
        ncv_out + (size_t)l * BATCH * NG * LEFT * HD);

    dim3 ag(SEQ / 16, NH, BATCH);
    attn_kernel<<<ag, 32, attn_lds, stream>>>(
        qbf, kfull, vfullt,
        rel_time + (size_t)l * (2 * MT - 1) * NH,
        rel_freq + (size_t)l * (2 * MF - 1) * NH, obf);

    // h = h + res_a[l] * (o @ Wo + bo)
    gemm(obf, DMODEL, WoT[l], DMODEL, bo + l * DMODEL, h, res_a, l,
         h, nullptr, DMODEL, Mrows, DMODEL, DMODEL);

    ln_kernel<<<Mrows, 256, 0, stream>>>(h, ln2_g + l * DMODEL, ln2_b + l * DMODEL, xinb);
    gemm(xinb, DMODEL, W1T[l], DMODEL, b1 + l * FFH, nullptr, nullptr, 0,
         gbuf1, nullptr, FFH, Mrows, FFH, DMODEL);
    gemm(xinb, DMODEL, W3T[l], DMODEL, b3 + l * FFH, nullptr, nullptr, 0,
         gbuf3, nullptr, FFH, Mrows, FFH, DMODEL);
    swiglu_kernel<<<(Mrows * FFH) / 256, 256, 0, stream>>>(gbuf1, gbuf3, ffbf, Mrows * FFH);
    // h = h + res_f[l] * (ff @ W2 + b2); last layer also emits bf16 h
    gemm(ffbf, FFH, W2T[l], FFH, b2 + l * DMODEL, h, res_f, l,
         h, (l == NL - 1) ? hbf : nullptr, DMODEL, Mrows, DMODEL, FFH);
  }

  gemm(hbf, DMODEL, outWT, DMODEL, out_b, nullptr, nullptr, 0,
       ytmp, nullptr, PATCH, Mrows, PATCH, DMODEL);
  unpatchify_kernel<<<(BATCH * CH * TDIM * FDIM) / 256, 256, 0, stream>>>(ytmp, y_out);

  (void)in_sizes; (void)n_in; (void)out_size; (void)ws_size;
}
